// sphere_plusLoss_36378372997788
// MI455X (gfx1250) — matrix operand order repaired and checked by reading, not compiled, not run on hardware
//
#include <hip/hip_runtime.h>
#include <hip/hip_bf16.h>

typedef __attribute__((ext_vector_type(16))) _Float16 v16h;
typedef __attribute__((ext_vector_type(8)))  float    v8f;

#define EDIM 512

// ---------------- K0: init workspace ----------------
__global__ void k_init(int* cnt, float* s_all, float* s_lab, float* scal, int C) {
    int i = blockIdx.x * blockDim.x + threadIdx.x;
    if (i < C)   cnt[i] = 0;
    if (i < EDIM) { s_all[i] = 0.0f; s_lab[i] = 0.0f; }
    if (i < 2)   scal[i] = 0.0f;
}

// ---------------- K1: histogram the labels ----------------
__global__ void k_count(const int* __restrict__ label, int* cnt, int B) {
    int i = blockIdx.x * blockDim.x + threadIdx.x;
    if (i < B) atomicAdd(&cnt[label[i]], 1);
}

// ---------------- K2: per-column sumsq -> rnorm, weights, sq_all/sq_lab ----
// Thread per column; wave reads 128B contiguous per row iteration (coalesced).
__global__ void k_colnorm(const float* __restrict__ K, const int* __restrict__ cnt,
                          float2* __restrict__ rw, float* scal, int C) {
    int c = blockIdx.x * blockDim.x + threadIdx.x;
    float t_all = 0.0f, t_lab = 0.0f;
    if (c < C) {
        const float* p = K + c;
        float a0 = 0.f, a1 = 0.f, a2 = 0.f, a3 = 0.f;
        for (int e = 0; e < EDIM; e += 4) {
            if (e + 32 < EDIM)
                __builtin_prefetch(p + (size_t)(e + 32) * C, 0, 0);
            float v0 = p[(size_t)(e + 0) * C];
            float v1 = p[(size_t)(e + 1) * C];
            float v2 = p[(size_t)(e + 2) * C];
            float v3 = p[(size_t)(e + 3) * C];
            a0 = fmaf(v0, v0, a0);
            a1 = fmaf(v1, v1, a1);
            a2 = fmaf(v2, v2, a2);
            a3 = fmaf(v3, v3, a3);
        }
        float ss = (a0 + a1) + (a2 + a3);
        float r  = rsqrtf(ss);
        float n  = (float)cnt[c];
        rw[c] = make_float2(r, n * r);
        t_all = ss * r * r;        // ~1.0 per column, computed numerically
        t_lab = n * t_all;
    }
    // wave32 reduction, then one atomic per wave
    for (int off = 16; off > 0; off >>= 1) {
        t_all += __shfl_xor(t_all, off);
        t_lab += __shfl_xor(t_lab, off);
    }
    if ((threadIdx.x & 31) == 0) {
        atomicAdd(&scal[0], t_all);
        atomicAdd(&scal[1], t_lab);
    }
}

// ---------------- K3: weighted row sums s_all[e], s_lab[e] ----------------
// One wave per (row e, column-chunk). rw[] (1.2MB total) stays in L2.
__global__ void k_rowsum(const float* __restrict__ K, const float2* __restrict__ rw,
                         float* s_all, float* s_lab, int C) {
    const int CPR = 8;  // chunks per row; grid = EDIM blocks * 8 waves
    int wid  = (blockIdx.x * blockDim.x + threadIdx.x) >> 5;
    int lane = threadIdx.x & 31;
    int e     = wid >> 3;
    int chunk = wid & (CPR - 1);
    if (e >= EDIM) return;
    int clen = (C + CPR - 1) / CPR;
    int c0 = chunk * clen;
    int c1 = min(C, c0 + clen);
    const float* rowp = K + (size_t)e * C;
    float aall = 0.0f, alab = 0.0f;
    for (int c = c0 + lane; c < c1; c += 32) {
        if (c + 256 < c1) __builtin_prefetch(rowp + c + 256, 0, 0);
        float  v = rowp[c];
        float2 w = rw[c];
        aall = fmaf(v, w.x, aall);
        alab = fmaf(v, w.y, alab);
    }
    for (int off = 16; off > 0; off >>= 1) {
        aall += __shfl_xor(aall, off);
        alab += __shfl_xor(alab, off);
    }
    if (lane == 0) {
        atomicAdd(&s_all[e], aall);
        atomicAdd(&s_lab[e], alab);
    }
}

// ---------------- K4: cross = s_lab . s_all via ONE v_wmma, final loss -----
// A[i][k] = s_lab[32i+k] (16x32), B[k][j] = s_all[32j+k] (32x16):
// D[i][j] = sum_k s_lab[32i+k]*s_all[32j+k]  =>  trace(D) = full 512-dot.
// Launched as exactly one wave32 (EXEC all ones).
__global__ void k_final(const float* __restrict__ s_all, const float* __restrict__ s_lab,
                        const float* __restrict__ scal, float* out, int C, int B) {
    int l  = threadIdx.x;      // 0..31
    int m  = l & 15;
    int hi = l >> 4;

    // 16-bit A 16x32 layout: lanes 0-15: h[0..7]=K 0..7, h[8..15]=K 16..23
    //                        lanes 16-31: h[0..7]=K 8..15, h[8..15]=K 24..31
    v16h a, b;
#pragma unroll
    for (int t = 0; t < 8; ++t) {
        a[t]     = (_Float16)s_lab[32 * m + 8 * hi + t];
        a[t + 8] = (_Float16)s_lab[32 * m + 16 + 8 * hi + t];
    }
    // 16-bit B 32x16 layout: lanes 0-15: h[0..7]=K 0..7, h[8..15]=K 16..23
    //                        lanes 16-31: h[0..7]=K 8..15, h[8..15]=K 24..31
#pragma unroll
    for (int t = 0; t < 8; ++t) {
        b[t]     = (_Float16)s_all[32 * m + 8 * hi + t];
        b[t + 8] = (_Float16)s_all[32 * m + 16 + 8 * hi + t];
    }

    v8f acc = {};
    acc = __builtin_amdgcn_wmma_f32_16x16x32_f16(
        /*neg_a=*/false, a, /*neg_b=*/false, b,
        /*c_mod=*/(short)0, acc, /*reuse_a=*/false, /*reuse_b=*/false);

    // Diagonal of D: lanes 0-7 hold D[l][l] in acc[l];
    //               lanes 24-31 hold D[l-16][l-16] in acc[l-24].
    float d = 0.0f;
#pragma unroll
    for (int k = 0; k < 8; ++k) {
        float v = acc[k];
        d += (l == k || l == k + 24) ? v : 0.0f;
    }
    for (int off = 16; off > 0; off >>= 1) d += __shfl_xor(d, off);

    if (l == 0) {
        float cross  = d;
        float sq_all = scal[0];
        float sq_lab = scal[1];
        // ALPHA == 1.0
        float loss = ((float)C * sq_lab + (float)B * sq_all - 2.0f * cross) / (float)C;
        out[0] = loss;
    }
}

extern "C" void kernel_launch(void* const* d_in, const int* in_sizes, int n_in,
                              void* d_out, int out_size, void* d_ws, size_t ws_size,
                              hipStream_t stream) {
    // inputs: embeddings (unused), kernel (E x C f32), label (B int)
    const float* Kmat  = (const float*)d_in[1];
    const int*   label = (const int*)d_in[2];
    const int E = EDIM;
    const int C = in_sizes[1] / E;
    const int B = in_sizes[2];

    // workspace layout
    char*   ws    = (char*)d_ws;
    float2* rw    = (float2*)ws;                       // 8*C bytes
    int*    cnt   = (int*)(ws + (size_t)8 * C);        // 4*C bytes
    float*  s_all = (float*)(ws + (size_t)12 * C);     // 512 f
    float*  s_lab = s_all + EDIM;                      // 512 f
    float*  scal  = s_lab + EDIM;                      // [0]=sq_all [1]=sq_lab

    int nbC = (C + 255) / 256;
    k_init   <<<nbC, 256, 0, stream>>>(cnt, s_all, s_lab, scal, C);
    k_count  <<<(B + 255) / 256, 256, 0, stream>>>(label, cnt, B);
    k_colnorm<<<nbC, 256, 0, stream>>>(Kmat, cnt, rw, scal, C);
    k_rowsum <<<E, 256, 0, stream>>>(Kmat, rw, s_all, s_lab, C);   // 512 blocks x 8 waves
    k_final  <<<1, 32, 0, stream>>>(s_all, s_lab, scal, (float*)d_out, C, B);
}
